// CrossLocalAttention_7851200217269
// MI455X (gfx1250) — matrix operand order repaired and checked by reading, not compiled, not run on hardware
//
#include <hip/hip_runtime.h>
#include <hip/hip_bf16.h>

typedef __attribute__((ext_vector_type(16))) __bf16 v16bf;
typedef __attribute__((ext_vector_type(8)))  __bf16 v8bf;
typedef __attribute__((ext_vector_type(8)))  float  v8f;

#define DIM   256
#define HEADS 8
#define HD    32
#define WIN   7
#define NTOK  49
#define IMG   112
#define HWSZ  (IMG*IMG)

// LDS layout (bf16 element offsets). Row strides padded to dodge bank conflicts.
#define SQK_STRIDE 264              // 256 + 8
#define SVT_STRIDE 72               // 64 + 8
#define SP_STRIDE  72

#define SQ_OFF   0
#define SK_OFF   (SQ_OFF  + 64*SQK_STRIDE)
#define SVT_OFF  (SK_OFF  + 64*SQK_STRIDE)
#define SO_OFF   (SVT_OFF + 256*SVT_STRIDE)
#define SP_OFF   (SO_OFF  + 64*SQK_STRIDE)
#define LDS_ELEMS (SP_OFF + 8*16*SP_STRIDE)
#define LDS_BYTES (LDS_ELEMS*2)

union BF16x16 { v16bf v; v8bf h[2]; };

__device__ __forceinline__ v8f wmma_bf16(v16bf a, v16bf b, v8f c) {
    return __builtin_amdgcn_wmma_f32_16x16x32_bf16(false, a, false, b, (short)0, c,
                                                   false, false);
}

// A-fragment (16x32 bf16) from LDS: lane half selects k chunks {+0..7,+16..23} / {+8..15,+24..31}
__device__ __forceinline__ v16bf ldsA(const __bf16* s, int stride, int lane, int row0, int k0) {
    BF16x16 r;
    const int half = lane >> 4;
    const __bf16* p = s + (row0 + (lane & 15)) * stride + k0 + half * 8;
    r.h[0] = *(const v8bf*)p;
    r.h[1] = *(const v8bf*)(p + 16);
    return r.v;
}

// B-fragment (32x16 bf16, stored transposed N x K row-major): lane = column N,
// lane half selects k chunks {+0..7,+16..23} / {+8..15,+24..31}.
__device__ __forceinline__ v16bf ldsB(const __bf16* s, int stride, int lane, int n0, int k0) {
    BF16x16 r;
    const int half = lane >> 4;
    const __bf16* p = s + (n0 + (lane & 15)) * stride + k0 + half * 8;
    r.h[0] = *(const v8bf*)p;
    r.h[1] = *(const v8bf*)(p + 16);
    return r.v;
}

// B-fragment from bf16 weight matrix in global memory, torch layout W[c_out][k] (= B^T).
__device__ __forceinline__ v16bf gW(const __bf16* w, int lane, int n0, int k0) {
    BF16x16 r;
    const int half = lane >> 4;
    const __bf16* p = w + (size_t)(n0 + (lane & 15)) * DIM + k0 + half * 8;
    r.h[0] = *(const v8bf*)p;
    r.h[1] = *(const v8bf*)(p + 16);
    return r.v;
}

// A-fragment gathered directly from the (B,C,H,W) fp32 image; img pre-offset to
// (b, 0, wh*7, ww*7). Rows (tokens) >= 49 are zero-filled.
__device__ __forceinline__ v16bf gA(const float* __restrict__ img, int lane, int m0, int k0) {
    BF16x16 r;
    const int half = lane >> 4;
    const int m = m0 + (lane & 15);
    if (m < NTOK) {
        const int rr = m / WIN, cc = m % WIN;
        const float* p = img + (size_t)(k0 + half * 8) * HWSZ + rr * IMG + cc;
#pragma unroll
        for (int i = 0; i < 8; ++i) {
            r.h[0][i] = (__bf16)p[(size_t)i * HWSZ];
            r.h[1][i] = (__bf16)p[(size_t)(i + 16) * HWSZ];
        }
    } else {
#pragma unroll
        for (int i = 0; i < 8; ++i) { r.h[0][i] = (__bf16)0.0f; r.h[1][i] = (__bf16)0.0f; }
    }
    return r.v;
}

__global__ void __launch_bounds__(256, 1) convert_weights_k(
    const float* __restrict__ wq, const float* __restrict__ wk,
    const float* __restrict__ wv, const float* __restrict__ wo,
    __bf16* __restrict__ o) {
    const int i = blockIdx.x * 256 + threadIdx.x;          // 0..65535
    o[i]          = (__bf16)wq[i];
    o[65536 + i]  = (__bf16)wk[i];
    o[131072 + i] = (__bf16)wv[i];
    o[196608 + i] = (__bf16)wo[i];
}

__global__ void __launch_bounds__(256, 1) cross_win_attn(
    const float* __restrict__ x, const float* __restrict__ y,
    const __bf16* __restrict__ wq, const __bf16* __restrict__ wk,
    const __bf16* __restrict__ wv, const __bf16* __restrict__ wo,
    const float* __restrict__ bq, const float* __restrict__ bk,
    const float* __restrict__ bv, const float* __restrict__ bo,
    float* __restrict__ out) {
    extern __shared__ __bf16 smem[];
    __bf16* sQ  = smem + SQ_OFF;
    __bf16* sK  = smem + SK_OFF;
    __bf16* sVt = smem + SVT_OFF;   // [256][SVT_STRIDE] : V transposed (d, j)
    __bf16* sO  = smem + SO_OFF;

    const int tid  = threadIdx.x;
    const int lane = tid & 31;
    const int wave = tid >> 5;
    const int half = lane >> 4;
    const int l15  = lane & 15;

    const int bid = blockIdx.x;          // 2048 windows: b(8) x wh(16) x ww(16)
    const int b   = bid >> 8;
    const int wh  = (bid >> 4) & 15;
    const int ww  = bid & 15;

    const float* xw = x + (size_t)b * DIM * HWSZ + (wh * WIN) * IMG + (ww * WIN);
    const float* yw = y + (size_t)b * DIM * HWSZ + (wh * WIN) * IMG + (ww * WIN);

    const int mi  = wave >> 1;           // row-tile 0..3
    const int njb = (wave & 1) * 8;      // col-tile base 0 or 8

    // ---------------- Phase 1a: Q = Xw @ Wq^T + bq -> sQ (bf16) ----------------
    // Software pipelined: B fragment double-buffered one tile ahead, A fragment
    // gathered one full k-step ahead so WMMAs never wait with loadcnt==0.
    {
        v8f acc[8];
#pragma unroll
        for (int j = 0; j < 8; ++j) {
            const float bias = bq[(njb + j) * 16 + l15];
#pragma unroll
            for (int r = 0; r < 8; ++r) acc[j][r] = bias;
        }
        v16bf a  = gA(xw, lane, mi * 16, 0);
        v16bf bc = gW(wq, lane, njb * 16, 0);
#pragma unroll 1
        for (int ks = 0; ks < 8; ++ks) {
            v16bf an = a;
            if (ks < 7) an = gA(xw, lane, mi * 16, ks * 32 + 32);
#pragma unroll
            for (int j = 0; j < 8; ++j) {
                const int jn  = (j + 1) & 7;
                const int ksn = (j == 7) ? ((ks + 1) & 7) : ks;
                const v16bf bn = gW(wq, lane, (njb + jn) * 16, ksn * 32);
                acc[j] = wmma_bf16(a, bc, acc[j]);
                bc = bn;
            }
            a = an;
        }
#pragma unroll
        for (int j = 0; j < 8; ++j) {
            const int col = (njb + j) * 16 + l15;
#pragma unroll
            for (int r = 0; r < 8; ++r)
                sQ[(mi * 16 + r + half * 8) * SQK_STRIDE + col] = (__bf16)acc[j][r];
        }
    }

    // ---------------- Phase 1b: K,V from y -> sK row-major, sVt transposed ------
    {
        v8f accK[8], accV[8];
#pragma unroll
        for (int j = 0; j < 8; ++j) {
            const float biK = bk[(njb + j) * 16 + l15];
            const float biV = bv[(njb + j) * 16 + l15];
#pragma unroll
            for (int r = 0; r < 8; ++r) { accK[j][r] = biK; accV[j][r] = biV; }
        }
        v16bf a   = gA(yw, lane, mi * 16, 0);
        v16bf bKc = gW(wk, lane, njb * 16, 0);
        v16bf bVc = gW(wv, lane, njb * 16, 0);
#pragma unroll 1
        for (int ks = 0; ks < 8; ++ks) {
            v16bf an = a;
            if (ks < 7) an = gA(yw, lane, mi * 16, ks * 32 + 32);
#pragma unroll
            for (int j = 0; j < 8; ++j) {
                const int jn  = (j + 1) & 7;
                const int ksn = (j == 7) ? ((ks + 1) & 7) : ks;
                const v16bf bKn = gW(wk, lane, (njb + jn) * 16, ksn * 32);
                const v16bf bVn = gW(wv, lane, (njb + jn) * 16, ksn * 32);
                accK[j] = wmma_bf16(a, bKc, accK[j]);
                accV[j] = wmma_bf16(a, bVc, accV[j]);
                bKc = bKn;
                bVc = bVn;
            }
            a = an;
        }
#pragma unroll
        for (int j = 0; j < 8; ++j) {
            const int col = (njb + j) * 16 + l15;
#pragma unroll
            for (int r = 0; r < 8; ++r) {
                const int row = mi * 16 + r + half * 8;
                sK[row * SQK_STRIDE + col] = (__bf16)accK[j][r];
                sVt[col * SVT_STRIDE + row] = (__bf16)accV[j][r];
            }
        }
    }
    __syncthreads();

    // ---------------- Phase 2: attention, one head per wave ---------------------
    {
        const int h = wave;
        const float scale = 0.17677669529663687f;   // 1/sqrt(32)
        __bf16* sP = smem + SP_OFF + wave * 16 * SP_STRIDE;

#pragma unroll 1
        for (int ti = 0; ti < 4; ++ti) {
            const int i0 = ti * 16;
            const v16bf aq = ldsA(sQ, SQK_STRIDE, lane, i0, h * HD);
            // preload all 4 K fragments, then issue the 4 WMMAs back-to-back
            v16bf kb[4];
#pragma unroll
            for (int nj = 0; nj < 4; ++nj)
                kb[nj] = ldsB(sK, SQK_STRIDE, lane, nj * 16, h * HD);
            v8f s[4];
#pragma unroll
            for (int nj = 0; nj < 4; ++nj) {
                v8f z;
#pragma unroll
                for (int r = 0; r < 8; ++r) z[r] = 0.0f;
                s[nj] = wmma_bf16(aq, kb[nj], z);
            }
            // scale + mask invalid columns (j = 49..63 live in tile 3, lanes l15 != 0)
#pragma unroll
            for (int nj = 0; nj < 4; ++nj)
#pragma unroll
                for (int r = 0; r < 8; ++r) s[nj][r] *= scale;
            if (l15 != 0) {
#pragma unroll
                for (int r = 0; r < 8; ++r) s[3][r] = -1.0e30f;
            }
            // softmax over j within each row (rows live per (r, half); reduce over
            // 4 tiles in-register then across the 16-lane half via shfl_xor)
            float rsum[8];
#pragma unroll
            for (int r = 0; r < 8; ++r) {
                float m4 = fmaxf(fmaxf(s[0][r], s[1][r]), fmaxf(s[2][r], s[3][r]));
                m4 = fmaxf(m4, __shfl_xor(m4, 1));
                m4 = fmaxf(m4, __shfl_xor(m4, 2));
                m4 = fmaxf(m4, __shfl_xor(m4, 4));
                m4 = fmaxf(m4, __shfl_xor(m4, 8));
                float sum = 0.0f;
#pragma unroll
                for (int nj = 0; nj < 4; ++nj) {
                    const float e = __expf(s[nj][r] - m4);
                    s[nj][r] = e;
                    sum += e;
                }
                sum += __shfl_xor(sum, 1);
                sum += __shfl_xor(sum, 2);
                sum += __shfl_xor(sum, 4);
                sum += __shfl_xor(sum, 8);
                rsum[r] = sum;
            }
            // stage un-normalized P as bf16 in per-wave LDS tile (C-layout -> A-layout)
#pragma unroll
            for (int nj = 0; nj < 4; ++nj)
#pragma unroll
                for (int r = 0; r < 8; ++r)
                    sP[(r + half * 8) * SP_STRIDE + nj * 16 + l15] = (__bf16)s[nj][r];

            // O_tile = P @ V_h   (K = 64 -> two k-steps; N = 32 -> two tiles)
            v8f o0, o1;
#pragma unroll
            for (int r = 0; r < 8; ++r) { o0[r] = 0.0f; o1[r] = 0.0f; }
#pragma unroll
            for (int kj = 0; kj < 2; ++kj) {
                const v16bf ap = ldsA(sP, SP_STRIDE, lane, 0, kj * 32);
                const v16bf bv0 = ldsB(sVt + (h * HD) * SVT_STRIDE, SVT_STRIDE, lane, 0,  kj * 32);
                const v16bf bv1 = ldsB(sVt + (h * HD) * SVT_STRIDE, SVT_STRIDE, lane, 16, kj * 32);
                o0 = wmma_bf16(ap, bv0, o0);
                o1 = wmma_bf16(ap, bv1, o1);
            }
            // normalize by softmax denominator (same lane mapping as C-layout rows)
#pragma unroll
            for (int r = 0; r < 8; ++r) {
                const float inv = 1.0f / rsum[r];
                const int row = i0 + r + half * 8;
                sO[row * SQK_STRIDE + h * HD + l15]      = (__bf16)(o0[r] * inv);
                sO[row * SQK_STRIDE + h * HD + 16 + l15] = (__bf16)(o1[r] * inv);
            }
        }
    }
    __syncthreads();

    // ---------------- Phase 3: Out = O @ Wo^T + bo, scatter (window_reverse) ----
    {
        v8f acc[8];
#pragma unroll
        for (int j = 0; j < 8; ++j) {
            const float bias = bo[(njb + j) * 16 + l15];
#pragma unroll
            for (int r = 0; r < 8; ++r) acc[j][r] = bias;
        }
        v16bf a  = ldsA(sO, SQK_STRIDE, lane, mi * 16, 0);
        v16bf bc = gW(wo, lane, njb * 16, 0);
#pragma unroll 1
        for (int ks = 0; ks < 8; ++ks) {
            v16bf an = a;
            if (ks < 7) an = ldsA(sO, SQK_STRIDE, lane, mi * 16, ks * 32 + 32);
#pragma unroll
            for (int j = 0; j < 8; ++j) {
                const int jn  = (j + 1) & 7;
                const int ksn = (j == 7) ? ((ks + 1) & 7) : ks;
                const v16bf bn = gW(wo, lane, (njb + jn) * 16, ksn * 32);
                acc[j] = wmma_bf16(a, bc, acc[j]);
                bc = bn;
            }
            a = an;
        }
        float* ob = out + (size_t)b * DIM * HWSZ + (wh * WIN) * IMG + (ww * WIN);
#pragma unroll
        for (int j = 0; j < 8; ++j) {
            const int ch = (njb + j) * 16 + l15;
#pragma unroll
            for (int r = 0; r < 8; ++r) {
                const int m = mi * 16 + r + half * 8;
                if (m < NTOK) {
                    const int rr = m / WIN, cc = m % WIN;
                    ob[(size_t)ch * HWSZ + rr * IMG + cc] = acc[j][r];
                }
            }
        }
    }
}

extern "C" void kernel_launch(void* const* d_in, const int* in_sizes, int n_in,
                              void* d_out, int out_size, void* d_ws, size_t ws_size,
                              hipStream_t stream) {
    (void)in_sizes; (void)n_in; (void)out_size; (void)ws_size;
    const float* x  = (const float*)d_in[0];
    const float* y  = (const float*)d_in[1];
    const float* Wq = (const float*)d_in[2];
    const float* bq = (const float*)d_in[3];
    const float* Wk = (const float*)d_in[4];
    const float* bk = (const float*)d_in[5];
    const float* Wv = (const float*)d_in[6];
    const float* bv = (const float*)d_in[7];
    const float* Wo = (const float*)d_in[8];
    const float* bo = (const float*)d_in[9];

    __bf16* wbf = (__bf16*)d_ws;   // 4 x 65536 bf16 = 512 KB

    convert_weights_k<<<256, 256, 0, stream>>>(Wq, Wk, Wv, Wo, wbf);

    hipFuncSetAttribute((const void*)cross_win_attn,
                        hipFuncAttributeMaxDynamicSharedMemorySize, LDS_BYTES);

    cross_win_attn<<<2048, 256, LDS_BYTES, stream>>>(
        x, y, wbf, wbf + 65536, wbf + 131072, wbf + 196608,
        bq, bk, bv, bo, (float*)d_out);
}
